// SelfAttention_71949292143146
// MI455X (gfx1250) — matrix operand order repaired and checked by reading, not compiled, not run on hardware
//
#include <hip/hip_runtime.h>
#include <hip/hip_bf16.h>

#define HIDDEN 1024
#define BATCH  8
#define SEQ    2048

// GEMM tiling
#define BM 128
#define BN 256
#define BK 64
#define LDS_ROW 72                 // 64 + 8 pad elements (16B) -> conflict-free
#define ASIZE (BM * LDS_ROW)       // 9216 elements
#define BSIZE (BN * LDS_ROW)       // 18432 elements
#define BUFSZ (ASIZE + BSIZE)      // 27648 elements per stage buffer

typedef __attribute__((ext_vector_type(16))) __bf16 v16bf;
typedef __attribute__((ext_vector_type(8)))  __bf16 v8bf;
typedef __attribute__((ext_vector_type(8)))  float  v8f;

// Exact parameter type expected by the async-to-LDS builtin (per hipcc diag):
// 'int __attribute__((vector_size(4*sizeof(int))))' pointers in AS1 / AS3.
typedef int v4i_vs __attribute__((vector_size(16)));
#define AS1 __attribute__((address_space(1)))
#define AS3 __attribute__((address_space(3)))

#if __has_builtin(__builtin_amdgcn_global_load_async_to_lds_b128)
#define ASYNC_LDS 1
#else
#define ASYNC_LDS 0
#endif

// 16B global -> LDS copy: async DMA (ASYNCcnt) when available, else sync.
__device__ __forceinline__ void copy16(__bf16* l, const __bf16* g) {
#if ASYNC_LDS
    __builtin_amdgcn_global_load_async_to_lds_b128(
        (AS1 v4i_vs*)g, (AS3 v4i_vs*)l, 0, 0);
#else
    *(v8bf*)l = *(const v8bf*)g;   // global_load_b128 + ds_store_b128
#endif
}

__device__ __forceinline__ void async_fence() {
#if ASYNC_LDS
#if __has_builtin(__builtin_amdgcn_s_wait_asynccnt)
    __builtin_amdgcn_s_wait_asynccnt(0);
#else
    asm volatile("s_wait_asynccnt 0x0" ::: "memory");
#endif
#endif
}

// Cooperative stage load: A tile BM x BK, B tile BN x BK into LDS buffer.
// Chunks of 8 bf16 (16B); consecutive threads hit consecutive 16B (coalesced).
__device__ __forceinline__ void stage_load(const __bf16* __restrict__ A,
                                           const __bf16* __restrict__ Bt,
                                           __bf16* sb, int mblk, int nblk,
                                           int K, int kk, int t) {
#pragma unroll
    for (int i = 0; i < 4; ++i) {            // 1024 A-chunks / 256 threads
        const int c = t + i * 256;
        const int row = c >> 3, col = (c & 7) * 8;
        copy16(sb + row * LDS_ROW + col,
               A + (size_t)(mblk + row) * K + kk + col);
    }
#pragma unroll
    for (int i = 0; i < 8; ++i) {            // 2048 B-chunks / 256 threads
        const int c = t + i * 256;
        const int row = c >> 3, col = (c & 7) * 8;
        copy16(sb + ASIZE + row * LDS_ROW + col,
               Bt + (size_t)(nblk + row) * K + kk + col);
    }
}

// ---------------------------------------------------------------------------
// LDS fragment loaders (CDNA5 WMMA VGPR layouts, CDNA5 ISA §7.12.2)
// A 16x32 bf16: lane<16: M=lane, K={0..7,16..23}; lane>=16: M=lane-16, K=+8.
// B 32x16 bf16 (fed as B^T rows): lane<16: N=lane, K={0..7,16..23}; lane>=16: K=+8.
// ---------------------------------------------------------------------------
__device__ __forceinline__ v16bf lds_frag_a(const __bf16* base, int mt, int kc) {
    const int lane = threadIdx.x & 31;
    const __bf16* p = base + (mt * 16 + (lane & 15)) * LDS_ROW + kc +
                      ((lane >> 4) << 3);
    union { v16bf v; v8bf h[2]; } u;
    u.h[0] = *(const v8bf*)(p);
    u.h[1] = *(const v8bf*)(p + 16);
    return u.v;
}

__device__ __forceinline__ v16bf lds_frag_b(const __bf16* base, int nt, int kc) {
    const int lane = threadIdx.x & 31;
    const __bf16* p = base + (nt * 16 + (lane & 15)) * LDS_ROW + kc +
                      ((lane >> 4) << 3);
    union { v16bf v; v8bf h[2]; } u;
    u.h[0] = *(const v8bf*)(p);
    u.h[1] = *(const v8bf*)(p + 16);
    return u.v;
}

// ---------------------------------------------------------------------------
// bf16 WMMA GEMM:  C[M,N] = alpha * (A[M,K] @ Bt[N,K]^T) + bias[n]
//   MODE 0: f32 row-major out; MODE 1: bf16 row-major out;
//   MODE 2: bf16 transposed V store  Cb[b*HIDDEN+n][s], m = b*SEQ+s
// 256 threads = 8 waves (2M x 4N), wave tile 64x64 (16 acc tiles = 128 VGPR),
// block tile 128x256, BK=64 double-buffered in LDS with async-to-LDS staging.
// ---------------------------------------------------------------------------
template <int MODE>
__global__ void __launch_bounds__(256)
gemm_bf16_wmma(const __bf16* __restrict__ A, const __bf16* __restrict__ Bt,
               float* __restrict__ Cf, __bf16* __restrict__ Cb,
               const float* __restrict__ bias,
               int M, int N, int K, float alpha,
               size_t strA, size_t strB, size_t strC) {
    __shared__ __bf16 smem[2 * BUFSZ];   // ~108 KB of the 320 KB WGP pool

    A  += (size_t)blockIdx.z * strA;
    Bt += (size_t)blockIdx.z * strB;

    const int t = threadIdx.x;
    const int wave = t >> 5;
    const int lane = t & 31;
    const int mblk = blockIdx.y * BM;
    const int nblk = blockIdx.x * BN;
    const int wmt = (wave >> 2) * 4;     // wave's first M-tile index (of 16x16)
    const int wnt = (wave & 3) * 4;      // wave's first N-tile index

    const v8f vzero = {0.f, 0.f, 0.f, 0.f, 0.f, 0.f, 0.f, 0.f};
    v8f acc[4][4];
#pragma unroll
    for (int i = 0; i < 4; ++i)
#pragma unroll
        for (int j = 0; j < 4; ++j) acc[i][j] = vzero;

    const int nstage = K / BK;

    stage_load(A, Bt, smem, mblk, nblk, K, 0, t);
    async_fence();
    __syncthreads();

    for (int s = 0; s < nstage; ++s) {
        const int buf = s & 1;
        if (s + 1 < nstage)       // overlap next-stage DMA with this compute
            stage_load(A, Bt, smem + (buf ^ 1) * BUFSZ, mblk, nblk, K,
                       (s + 1) * BK, t);

        const __bf16* sa = smem + buf * BUFSZ;
        const __bf16* sbb = sa + ASIZE;

#pragma unroll
        for (int kc = 0; kc < BK; kc += 32) {
            v16bf af[4], bfr[4];
#pragma unroll
            for (int i = 0; i < 4; ++i) af[i] = lds_frag_a(sa, wmt + i, kc);
#pragma unroll
            for (int j = 0; j < 4; ++j) bfr[j] = lds_frag_b(sbb, wnt + j, kc);
#pragma unroll
            for (int i = 0; i < 4; ++i)
#pragma unroll
                for (int j = 0; j < 4; ++j)
                    acc[i][j] = __builtin_amdgcn_wmma_f32_16x16x32_bf16(
                        false, af[i], false, bfr[j], (short)0, acc[i][j],
                        false, false);
        }

        if (s + 1 < nstage) {
            async_fence();       // my wave's async writes to buf^1 landed
            __syncthreads();     // everyone's landed; buf^1 ready, buf free
        }
    }

    // Epilogue. C/D layout: lane<16 -> N=lane, M=+r; lane>=16 -> M=+8+r.
    const int cn = lane & 15;
    const int cm = (lane >> 4) << 3;
#pragma unroll
    for (int i = 0; i < 4; ++i) {
#pragma unroll
        for (int j = 0; j < 4; ++j) {
            const int n  = nblk + (wnt + j) * 16 + cn;
            const int mb = mblk + (wmt + i) * 16 + cm;
            const float bb = (bias != nullptr) ? bias[n] : 0.f;
#pragma unroll
            for (int r = 0; r < 8; ++r) {
                const float val = acc[i][j][r] * alpha + bb;
                if (MODE == 0) {
                    Cf[(size_t)blockIdx.z * strC + (size_t)(mb + r) * N + n] = val;
                } else if (MODE == 1) {
                    Cb[(size_t)blockIdx.z * strC + (size_t)(mb + r) * N + n] =
                        (__bf16)val;
                } else {  // MODE 2: Vt[b][n][s]
                    const int m = mb + r;
                    const int b = m >> 11;
                    const int sS = m & (SEQ - 1);
                    Cb[((size_t)b * HIDDEN + n) * SEQ + sS] = (__bf16)val;
                }
            }
        }
    }
}

// ---------------------------------------------------------------------------
// In-place row softmax over SEQ=2048 bf16 values, fp32 math.
// ---------------------------------------------------------------------------
__global__ void __launch_bounds__(256) softmax_rows(__bf16* __restrict__ S) {
    __bf16* p = S + (size_t)blockIdx.x * SEQ;
    const int t = threadIdx.x;
    const int wave = t >> 5, lane = t & 31;

    float v[8];
    float mx = -3.4e38f;
#pragma unroll
    for (int i = 0; i < 8; ++i) {
        v[i] = (float)p[t + i * 256];
        mx = fmaxf(mx, v[i]);
    }
#pragma unroll
    for (int off = 16; off > 0; off >>= 1)
        mx = fmaxf(mx, __shfl_xor(mx, off, 32));

    __shared__ float redm[8];
    __shared__ float reds[8];
    if (lane == 0) redm[wave] = mx;
    __syncthreads();
#pragma unroll
    for (int w = 0; w < 8; ++w) mx = fmaxf(mx, redm[w]);

    float sum = 0.f;
#pragma unroll
    for (int i = 0; i < 8; ++i) {
        v[i] = __expf(v[i] - mx);
        sum += v[i];
    }
#pragma unroll
    for (int off = 16; off > 0; off >>= 1) sum += __shfl_xor(sum, off, 32);
    if (lane == 0) reds[wave] = sum;
    __syncthreads();
    sum = 0.f;
#pragma unroll
    for (int w = 0; w < 8; ++w) sum += reds[w];

    const float inv = 1.0f / sum;
#pragma unroll
    for (int i = 0; i < 8; ++i) p[t + i * 256] = (__bf16)(v[i] * inv);
}

// ---------------------------------------------------------------------------
// Conversions
// ---------------------------------------------------------------------------
__global__ void __launch_bounds__(256)
cvt_f32_to_bf16(const float* __restrict__ x, __bf16* __restrict__ y, int n) {
    const int i = blockIdx.x * 256 + threadIdx.x;
    if (i < n) y[i] = (__bf16)x[i];
}

__global__ void __launch_bounds__(256)
cvt_w_transpose(const float* __restrict__ W, __bf16* __restrict__ Wt) {
    const int idx = blockIdx.x * 256 + threadIdx.x;
    const int o = idx >> 10;
    const int i = idx & (HIDDEN - 1);
    Wt[idx] = (__bf16)W[((size_t)i << 10) + o];
}

// ---------------------------------------------------------------------------
extern "C" void kernel_launch(void* const* d_in, const int* in_sizes, int n_in,
                              void* d_out, int out_size, void* d_ws,
                              size_t ws_size, hipStream_t stream) {
    (void)in_sizes; (void)n_in; (void)out_size; (void)ws_size;
    const float* x  = (const float*)d_in[0];
    const float* Wq = (const float*)d_in[1];
    const float* bq = (const float*)d_in[2];
    const float* Wk = (const float*)d_in[3];
    const float* bk = (const float*)d_in[4];
    const float* Wv = (const float*)d_in[5];
    const float* bv = (const float*)d_in[6];
    float* out = (float*)d_out;

    const size_t MTOT = (size_t)BATCH * SEQ;  // 16384

    char* ws = (char*)d_ws;
    __bf16* xb  = (__bf16*)ws; ws += MTOT * HIDDEN * 2;             // 32 MB
    __bf16* Wqt = (__bf16*)ws; ws += (size_t)HIDDEN * HIDDEN * 2;   // 2 MB
    __bf16* Wkt = (__bf16*)ws; ws += (size_t)HIDDEN * HIDDEN * 2;
    __bf16* Wvt = (__bf16*)ws; ws += (size_t)HIDDEN * HIDDEN * 2;
    __bf16* Qb  = (__bf16*)ws; ws += MTOT * HIDDEN * 2;             // 32 MB
    __bf16* Kb  = (__bf16*)ws; ws += MTOT * HIDDEN * 2;             // 32 MB
    __bf16* Vt  = (__bf16*)ws; ws += MTOT * HIDDEN * 2;             // 32 MB [B][D][S]
    __bf16* Sb  = (__bf16*)ws; ws += (size_t)BATCH * SEQ * SEQ * 2; // 64 MB [B][S][S]

    // 1) precision conversion
    const int nx = (int)(MTOT * HIDDEN);
    cvt_f32_to_bf16<<<nx / 256, 256, 0, stream>>>(x, xb, nx);
    const int nw = HIDDEN * HIDDEN;
    cvt_w_transpose<<<nw / 256, 256, 0, stream>>>(Wq, Wqt);
    cvt_w_transpose<<<nw / 256, 256, 0, stream>>>(Wk, Wkt);
    cvt_w_transpose<<<nw / 256, 256, 0, stream>>>(Wv, Wvt);

    // 2) QKV projections
    dim3 gQKV(HIDDEN / BN, (unsigned)(MTOT / BM), 1);
    gemm_bf16_wmma<1><<<gQKV, 256, 0, stream>>>(
        xb, Wqt, nullptr, Qb, bq, (int)MTOT, HIDDEN, HIDDEN, 1.0f, 0, 0, 0);
    gemm_bf16_wmma<1><<<gQKV, 256, 0, stream>>>(
        xb, Wkt, nullptr, Kb, bk, (int)MTOT, HIDDEN, HIDDEN, 1.0f, 0, 0, 0);
    gemm_bf16_wmma<2><<<gQKV, 256, 0, stream>>>(
        xb, Wvt, nullptr, Vt, bv, (int)MTOT, HIDDEN, HIDDEN, 1.0f, 0, 0, 0);

    // 3) scores: S[b] = (Q[b] @ K[b]^T) / 32
    dim3 gS(SEQ / BN, SEQ / BM, BATCH);
    gemm_bf16_wmma<1><<<gS, 256, 0, stream>>>(
        Qb, Kb, nullptr, Sb, nullptr, SEQ, SEQ, HIDDEN, 0.03125f,
        (size_t)SEQ * HIDDEN, (size_t)SEQ * HIDDEN, (size_t)SEQ * SEQ);

    // 4) softmax rows, in place
    softmax_rows<<<(unsigned)(BATCH * SEQ), 256, 0, stream>>>(Sb);

    // 5) output: O[b] = P[b] @ V[b]
    dim3 gO(HIDDEN / BN, SEQ / BM, BATCH);
    gemm_bf16_wmma<0><<<gO, 256, 0, stream>>>(
        Sb, Vt, out, nullptr, nullptr, SEQ, HIDDEN, SEQ, 1.0f,
        (size_t)SEQ * SEQ, (size_t)HIDDEN * SEQ, (size_t)SEQ * HIDDEN);
}
